// RelativePositionEncoder_2147483648136
// MI455X (gfx1250) — matrix operand order repaired and checked by reading, not compiled, not run on hardware
//
#include <hip/hip_runtime.h>
#include <hip/hip_bf16.h>

typedef __attribute__((ext_vector_type(16))) _Float16 v16h;
typedef __attribute__((ext_vector_type(2)))  _Float16 v2h;
typedef __attribute__((ext_vector_type(8)))  float    v8f;

#define HALF_DIM 64
#define H_STRIDE 68   // padded LDS row stride (halfs) to spread banks
#define WAVES_PER_BLOCK 8

// ---------------------------------------------------------------------------
// Edge kernel: one wave == one 16-edge tile (grid-stride over tile groups).
// Phase 1 (scalar): rel-vec normalize, 3->64 linear, LayerNorm, exact GELU.
// Phase 2 (matrix): h[16x64] @ W2[64x64] via 8x v_wmma_f32_16x16x32_f16.
// Phase 3 (scatter): straight-line global_atomic_add_f32 into agg[dst],
//   unsigned 32-bit indices so all atomics use the saddr+voffset scale_offset
//   addressing form. Tail edges are clamped to a dummy row (nNodes).
// ---------------------------------------------------------------------------
__global__ __launch_bounds__(256) void edge_encode_scatter_kernel(
    const float* __restrict__ pos,   // [N,3]
    const int*   __restrict__ ei,    // [2,E]
    const float* __restrict__ W1,    // [3,64]
    const float* __restrict__ b1,    // [64]
    const float* __restrict__ ln_g,  // [64]
    const float* __restrict__ ln_b,  // [64]
    const float* __restrict__ W2,    // [64,64]
    const float* __restrict__ b2,    // [64]
    float* __restrict__ agg,         // [N+1,64]  (last row = dummy)
    float* __restrict__ deg,         // [N+1]
    int nEdges, int nNodes, int nGroups)
{
    __shared__ _Float16 sW2[HALF_DIM * HALF_DIM];            // 8 KB, f16 copy of W2
    __shared__ float    sW1[3 * HALF_DIM];
    __shared__ float    sB1[HALF_DIM], sG[HALF_DIM], sB[HALF_DIM], sB2[HALF_DIM];
    __shared__ _Float16 sH[WAVES_PER_BLOCK][16 * H_STRIDE];  // per-wave h tile (f16)

    const int t   = threadIdx.x;
    const int lid = t & 31;          // lane in wave32
    const int wv  = t >> 5;          // wave id in block

    // -------- one-time block staging of parameters into LDS --------
    for (int i = t; i < HALF_DIM * HALF_DIM; i += 256)
        sW2[i] = (_Float16)W2[i];
    for (int i = t; i < 3 * HALF_DIM; i += 256)
        sW1[i] = W1[i];
    if (t < HALF_DIM) {
        sB1[t] = b1[t];
        sG[t]  = ln_g[t];
        sB[t]  = ln_b[t];
        sB2[t] = b2[t];
    }
    __syncthreads();

    // -------- load constant B fragments (W2) once per wave --------
    // B 32x16 f16 layout: lane: col = lid&15, K = (lid>>4)*16 + j
    const int bcol = lid & 15;
    const int bkb  = (lid >> 4) * 16;
    v16h bfrag[4][2];
#pragma unroll
    for (int n = 0; n < 4; ++n)
#pragma unroll
        for (int kc = 0; kc < 2; ++kc)
#pragma unroll
            for (int j = 0; j < 16; ++j)
                bfrag[n][kc][j] = sW2[(kc * 32 + bkb + j) * HALF_DIM + n * 16 + bcol];

    // A-fragment addressing constants (16x32 f16, contiguous K per lane half, same as B)
    const int arow  = lid & 15;
    const int khalf = lid >> 4;      // 0: K 0-15, 1: K 16-31
    const unsigned ocol  = (unsigned)(lid & 15);
    const int      rbase = (lid >> 4) * 8;

    for (int g = blockIdx.x; g < nGroups; g += gridDim.x) {
        const int tile = g * WAVES_PER_BLOCK + wv;
        const int e0   = tile * 16;
        if (e0 >= nEdges) continue;

        // ---------------- phase 1: per-edge scalar front end ----------------
        const int er = lid & 15;              // edge-in-tile (both halves mirror)
        const int e  = e0 + er;
        const bool ev = (e < nEdges);
        const int ec = ev ? e : (nEdges - 1);
        const int s  = ei[ec];
        const int d  = ei[nEdges + ec];
        const int dclamp = ev ? d : nNodes;   // tail edges -> dummy row
        if (lid < 16) unsafeAtomicAdd(&deg[(unsigned)dclamp], 1.0f);

        const float ax = pos[3 * s], ay = pos[3 * s + 1], az = pos[3 * s + 2];
        float rx = pos[3 * d]     - ax;
        float ry = pos[3 * d + 1] - ay;
        float rz = pos[3 * d + 2] - az;
        const float dist = sqrtf(rx * rx + ry * ry + rz * rz);
        const float inv  = 1.0f / (dist + 1e-6f);
        rx *= inv; ry *= inv; rz *= inv;

        // each half-wave covers 32 channels of this edge
        const int c0 = (lid >> 4) * 32;
        float hv[32];
        float sum = 0.f, sumsq = 0.f;
#pragma unroll
        for (int i = 0; i < 32; ++i) {
            const int c = c0 + i;
            const float v = rx * sW1[c] + ry * sW1[HALF_DIM + c]
                          + rz * sW1[2 * HALF_DIM + c] + sB1[c];
            hv[i] = v;
            sum += v; sumsq += v * v;
        }
        sum   += __shfl_xor(sum,   16, 32);
        sumsq += __shfl_xor(sumsq, 16, 32);
        const float mu   = sum * (1.0f / HALF_DIM);
        const float var  = sumsq * (1.0f / HALF_DIM) - mu * mu;
        const float rstd = rsqrtf(var + 1e-5f);

#pragma unroll
        for (int i = 0; i < 32; i += 2) {
            const int c = c0 + i;
            const float x0 = (hv[i]     - mu) * rstd * sG[c]     + sB[c];
            const float x1 = (hv[i + 1] - mu) * rstd * sG[c + 1] + sB[c + 1];
            const float g0 = 0.5f * x0 * (1.0f + erff(x0 * 0.70710678118654752f));
            const float g1 = 0.5f * x1 * (1.0f + erff(x1 * 0.70710678118654752f));
            v2h p; p.x = (_Float16)g0; p.y = (_Float16)g1;
            *reinterpret_cast<v2h*>(&sH[wv][er * H_STRIDE + c]) = p;
        }

        // per-lane scatter row offsets (edge rbase+r's dst row, pre-scaled):
        // every lane holds dclamp for edge lid&15, so source lanes 0..15 cover
        // all 16 edges; 8 ds_bpermute hoisted out of the N-tile loop.
        // Unsigned so atomics use 32-bit voffset + saddr scale_offset form.
        unsigned rowoff[8];
#pragma unroll
        for (int r = 0; r < 8; ++r)
            rowoff[r] = (unsigned)__shfl(dclamp, rbase + r, 32) * HALF_DIM;

        // wave-private LDS: stores precede transposed reads (DS is in-order);
        // fence blocks compiler reordering and covers counter readiness.
        asm volatile("s_wait_dscnt 0x0" ::: "memory");

        // ---------------- phase 2: A fragments + WMMA ----------------
        v16h afrag[2];
#pragma unroll
        for (int kc = 0; kc < 2; ++kc)
#pragma unroll
            for (int j = 0; j < 16; ++j) {
                const int K = kc * 32 + khalf * 16 + j;
                afrag[kc][j] = sH[wv][arow * H_STRIDE + K];
            }

        // ---------------- phase 3: 4 N-tiles, accumulate + scatter ----------
#pragma unroll
        for (int n = 0; n < 4; ++n) {
            v8f acc = {};
            acc = __builtin_amdgcn_wmma_f32_16x16x32_f16(
                false, afrag[0], false, bfrag[n][0], (short)0, acc, false, false);
            acc = __builtin_amdgcn_wmma_f32_16x16x32_f16(
                false, afrag[1], false, bfrag[n][1], (short)0, acc, false, false);
            const unsigned c  = (unsigned)(n * 16) + ocol;
            const float    bc = sB2[c];
#pragma unroll
            for (int r = 0; r < 8; ++r)
                unsafeAtomicAdd(&agg[rowoff[r] + c], acc[r] + bc);
        }
    }
}

// ---------------------------------------------------------------------------
// Finalize: out[n, 0:64] = agg[n] / max(deg[n], 1);  out[n, 64:128] = 0
// ---------------------------------------------------------------------------
__global__ __launch_bounds__(256) void finalize_kernel(
    const float* __restrict__ agg, const float* __restrict__ deg,
    float* __restrict__ out, int nNodes)
{
    const int i = blockIdx.x * blockDim.x + threadIdx.x;
    const int total = nNodes * HALF_DIM;
    if (i >= total) return;
    const unsigned n = (unsigned)i >> 6;
    const unsigned c = (unsigned)i & 63u;
    float dg = deg[n];
    dg = dg < 1.0f ? 1.0f : dg;
    out[n * 128 + c]       = agg[(unsigned)i] / dg;
    out[n * 128 + 64 + c]  = 0.0f;
}

extern "C" void kernel_launch(void* const* d_in, const int* in_sizes, int n_in,
                              void* d_out, int out_size, void* d_ws, size_t ws_size,
                              hipStream_t stream) {
    const float* pos  = (const float*)d_in[0];
    const int*   ei   = (const int*)d_in[1];
    // d_in[2] = batch (unused)
    const float* W1   = (const float*)d_in[3];
    const float* b1   = (const float*)d_in[4];
    const float* ln_g = (const float*)d_in[5];
    const float* ln_b = (const float*)d_in[6];
    const float* W2   = (const float*)d_in[7];
    const float* b2   = (const float*)d_in[8];

    const int nNodes = in_sizes[0] / 3;
    const int nEdges = in_sizes[1] / 2;

    float* agg = (float*)d_ws;                               // [(nNodes+1)*64]
    float* deg = agg + (size_t)(nNodes + 1) * HALF_DIM;      // [nNodes+1]
    hipMemsetAsync(d_ws, 0,
                   ((size_t)(nNodes + 1) * HALF_DIM + (size_t)(nNodes + 1))
                       * sizeof(float),
                   stream);

    const int nTiles  = (nEdges + 15) / 16;
    const int nGroups = (nTiles + WAVES_PER_BLOCK - 1) / WAVES_PER_BLOCK;
    const int blocks  = nGroups < 1280 ? nGroups : 1280;

    edge_encode_scatter_kernel<<<blocks, 256, 0, stream>>>(
        pos, ei, W1, b1, ln_g, ln_b, W2, b2, agg, deg, nEdges, nNodes, nGroups);

    const int fin = nNodes * HALF_DIM;
    finalize_kernel<<<(fin + 255) / 256, 256, 0, stream>>>(agg, deg,
                                                           (float*)d_out, nNodes);
}
